// ModelCordinateDescentAlpha_91164975824956
// MI455X (gfx1250) — matrix operand order repaired and checked by reading, not compiled, not run on hardware
//
#include <hip/hip_runtime.h>

#define N_ROWS    16384
#define DIM       512
#define N_CTR     4096
#define OUT_DIM   16
#define BM        64
#define WAVES     4
#define BLOCK     (WAVES * 32)
#define CHUNK     32
#define XS_STRIDE 528   // bytes/row = 1056, multiple of 32 -> aligned ds_load_b128
#define KT_STRIDE 48    // bytes/row = 96, multiple of 16

typedef __attribute__((ext_vector_type(16))) __bf16 v16bf;
typedef __attribute__((ext_vector_type(8)))  __bf16 v8bf;
typedef __attribute__((ext_vector_type(4)))  __bf16 v4bf;
typedef __attribute__((ext_vector_type(8)))  float  v8f;
typedef __attribute__((ext_vector_type(4)))  float  v4f;

union V16U { v16bf v; v8bf h[2]; };

// One wave per row: ||row||^2 reduction, lane-strided float4 loads + shfl_xor tree.
__global__ void __launch_bounds__(256) rownorm_kernel(const float* __restrict__ src,
                                                      float* __restrict__ dst) {
  const int row  = blockIdx.x * 8 + (threadIdx.x >> 5);
  const int lane = threadIdx.x & 31;
  const v4f* s4 = reinterpret_cast<const v4f*>(src) + (size_t)row * (DIM / 4);
  float sum = 0.f;
#pragma unroll
  for (int i = 0; i < DIM / 4 / 32; ++i) {
    v4f v = s4[lane + i * 32];
    sum += v.x * v.x + v.y * v.y + v.z * v.z + v.w * v.w;
  }
#pragma unroll
  for (int m = 16; m >= 1; m >>= 1) sum += __shfl_xor(sum, m, 32);
  if (lane == 0) dst[row] = sum;
}

// Fused: k = exp(gamma*(2 X Mu^T - |x|^2 - |mu|^2));  out = k @ alpha
__global__ void __launch_bounds__(BLOCK) rbf_head_kernel(
    const float* __restrict__ x, const float* __restrict__ mu,
    const float* __restrict__ gamma_p, const float* __restrict__ alpha,
    const float* __restrict__ dxs, const float* __restrict__ dmus,
    float* __restrict__ out) {
  __shared__ __attribute__((aligned(32))) __bf16 xs[BM][XS_STRIDE];
  __shared__ __attribute__((aligned(32))) __bf16 mus[CHUNK][XS_STRIDE];
  __shared__ __attribute__((aligned(64))) __bf16 alt[OUT_DIM][CHUNK];  // alpha^T
  __shared__ __attribute__((aligned(32))) __bf16 kts[WAVES][16][KT_STRIDE];

  const int tid  = threadIdx.x;
  const int lane = tid & 31;
  const int w    = tid >> 5;
  const int half = lane >> 4;   // ISA 16-bit A/B layouts split lanes 0-15 / 16-31
  const int l16  = lane & 15;
  const int rowBase = blockIdx.x * BM;

  const float g = gamma_p[0];

  // Stage x tile -> bf16 LDS (packed cvt + 8B stores), loaded exactly once.
  {
    const v4f* x4 = reinterpret_cast<const v4f*>(x + (size_t)rowBase * DIM);
#pragma unroll
    for (int it = 0; it < (BM * DIM / 4) / BLOCK; ++it) {
      int idx = tid + it * BLOCK;
      int r = idx >> 7;           // 128 float4 per row
      int c = (idx & 127) << 2;
      v4f v = x4[idx];
      *reinterpret_cast<v4bf*>(&xs[r][c]) = __builtin_convertvector(v, v4bf);
    }
  }
  __syncthreads();

  // Hoist this wave's row norms (C-layout: lane-half selects M+8).
  float dxv[8];
#pragma unroll
  for (int j = 0; j < 8; ++j)
    dxv[j] = dxs[rowBase + w * 16 + half * 8 + j];

  v8f outacc = {};

  for (int cc = 0; cc < N_CTR; cc += CHUNK) {
    __syncthreads();  // previous chunk fully consumed before restaging
    {
      const v4f* m4 = reinterpret_cast<const v4f*>(mu + (size_t)cc * DIM);
#pragma unroll
      for (int it = 0; it < (CHUNK * DIM / 4) / BLOCK; ++it) {
        int idx = tid + it * BLOCK;
        int r = idx >> 7;
        int c = (idx & 127) << 2;
        v4f v = m4[idx];
        *reinterpret_cast<v4bf*>(&mus[r][c]) = __builtin_convertvector(v, v4bf);
      }
      // alpha chunk: 32x16 f32, staged TRANSPOSED -> alt[out][center] so the
      // B-fragment read is two 16B v8bf row loads, 32B apart, per lane.
      const v4f* a4 = reinterpret_cast<const v4f*>(alpha + (size_t)cc * OUT_DIM);
      v4f v = a4[tid];
      int r = tid >> 2;           // center within chunk
      int c = (tid & 3) << 2;     // out-dim base
      v4bf b = __builtin_convertvector(v, v4bf);
      alt[c + 0][r] = b.x; alt[c + 1][r] = b.y;
      alt[c + 2][r] = b.z; alt[c + 3][r] = b.w;
    }
    __syncthreads();

    // Prefetch next mu chunk into cache (global_prefetch_b8); 1 line / thread / step.
    if (cc + CHUNK < N_CTR) {
      const float* nxt = mu + (size_t)(cc + CHUNK) * DIM;
#pragma unroll
      for (int p = 0; p < 4; ++p)
        __builtin_prefetch(nxt + (size_t)tid * 32 + (size_t)p * 4096, 0, 1);
    }

    v8f acc0 = {}, acc1 = {};
#pragma unroll 4
    for (int kk = 0; kk < DIM; kk += 32) {
      // A frag (16x32 bf16): lane half selects K base 0/8; two 16B chunks 32B apart.
      V16U a;
      const __bf16* ap = &xs[w * 16 + l16][kk + half * 8];
      a.h[0] = *reinterpret_cast<const v8bf*>(ap);
      a.h[1] = *reinterpret_cast<const v8bf*>(ap + 16);
      // B frags (32x16 bf16): lane = column (center); lane half selects K base 0/8; two 16B chunks 32B apart.
      V16U b0u, b1u;
      b0u.h[0] = *reinterpret_cast<const v8bf*>(&mus[l16][kk + half * 8]);
      b0u.h[1] = *reinterpret_cast<const v8bf*>(&mus[l16][kk + half * 8 + 16]);
      b1u.h[0] = *reinterpret_cast<const v8bf*>(&mus[16 + l16][kk + half * 8]);
      b1u.h[1] = *reinterpret_cast<const v8bf*>(&mus[16 + l16][kk + half * 8 + 16]);
      v16bf b0 = b0u.v;
      v16bf b1 = b1u.v;
      acc0 = __builtin_amdgcn_wmma_f32_16x16x32_bf16(false, a.v, false, b0,
                                                     (short)0, acc0, false, false);
      acc1 = __builtin_amdgcn_wmma_f32_16x16x32_bf16(false, a.v, false, b1,
                                                     (short)0, acc1, false, false);
    }

    // Epilogue: RBF kernel value, write 16x32 k-tile (bf16) via per-wave LDS
    // scratch to convert C-layout -> A-layout for the alpha GEMM.
    const float dmu0 = dmus[cc + l16];
    const float dmu1 = dmus[cc + 16 + l16];
#pragma unroll
    for (int j = 0; j < 8; ++j) {
      float e0 = expf(g * (2.f * acc0[j] - dxv[j] - dmu0));
      float e1 = expf(g * (2.f * acc1[j] - dxv[j] - dmu1));
      kts[w][half * 8 + j][l16]      = (__bf16)e0;
      kts[w][half * 8 + j][16 + l16] = (__bf16)e1;
    }
    // kfrag: A-layout 16(M rows) x 32(K centers)
    V16U kf;
    const __bf16* kp = &kts[w][l16][half * 8];
    kf.h[0] = *reinterpret_cast<const v8bf*>(kp);
    kf.h[1] = *reinterpret_cast<const v8bf*>(kp + 16);
    // alpha frag: B-layout 32(K centers) x 16(N out); from alt^T, K base 0/8 per lane half, two 16B chunks.
    V16U afu;
    afu.h[0] = *reinterpret_cast<const v8bf*>(&alt[l16][half * 8]);
    afu.h[1] = *reinterpret_cast<const v8bf*>(&alt[l16][half * 8 + 16]);
    v16bf af = afu.v;
    outacc = __builtin_amdgcn_wmma_f32_16x16x32_bf16(false, kf.v, false, af,
                                                     (short)0, outacc, false, false);
  }

  // Exclusive ownership of this row tile -> plain stores, deterministic.
#pragma unroll
  for (int j = 0; j < 8; ++j)
    out[(size_t)(rowBase + w * 16 + half * 8 + j) * OUT_DIM + l16] = outacc[j];
}

extern "C" void kernel_launch(void* const* d_in, const int* in_sizes, int n_in,
                              void* d_out, int out_size, void* d_ws, size_t ws_size,
                              hipStream_t stream) {
  (void)in_sizes; (void)n_in; (void)out_size; (void)ws_size;
  const float* x     = (const float*)d_in[0];
  const float* mu    = (const float*)d_in[1];
  const float* gamma = (const float*)d_in[2];
  const float* alpha = (const float*)d_in[3];
  float* out  = (float*)d_out;
  float* dmus = (float*)d_ws;        // [4096]  ||mu||^2
  float* dxs  = dmus + N_CTR;        // [16384] ||x||^2

  rownorm_kernel<<<N_CTR / 8, 256, 0, stream>>>(mu, dmus);
  rownorm_kernel<<<N_ROWS / 8, 256, 0, stream>>>(x, dxs);
  rbf_head_kernel<<<N_ROWS / BM, BLOCK, 0, stream>>>(x, mu, gamma, alpha, dxs, dmus, out);
}
